// MultiHeadCrossAttention_39350490366195
// MI455X (gfx1250) — matrix operand order repaired and checked by reading, not compiled, not run on hardware
//
#include <hip/hip_runtime.h>

typedef __bf16 bh;
typedef __attribute__((ext_vector_type(4)))  bh    bhx4;
typedef __attribute__((ext_vector_type(8)))  bh    bhx8;
typedef __attribute__((ext_vector_type(16))) bh    bhx16;
typedef __attribute__((ext_vector_type(8)))  float fx8;

#define SEQ    2048
#define DMODEL 1024
#define NHEAD  16
#define DHEAD  64
#define BATCH  2
#define MTOT   (BATCH*SEQ)   // 4096

__device__ __forceinline__ fx8 fzero8() {
  fx8 z = {0.f,0.f,0.f,0.f,0.f,0.f,0.f,0.f};
  return z;
}

__device__ __forceinline__ bhx16 cat8(bhx8 lo, bhx8 hi) {
  return __builtin_shufflevector(lo, hi, 0,1,2,3,4,5,6,7,8,9,10,11,12,13,14,15);
}

// CDNA5 async DMA: global -> LDS, 16 bytes per lane (ASYNCcnt-tracked).
// lds_off = wave-relative LDS byte offset (low 32 bits of generic shared ptr).
__device__ __forceinline__ void async_copy_b128(unsigned lds_off, const void* gptr) {
  asm volatile("global_load_async_to_lds_b128 %0, %1, off"
               :: "v"(lds_off), "v"(gptr) : "memory");
}
__device__ __forceinline__ void wait_async0() {
  asm volatile("s_wait_asynccnt 0x0" ::: "memory");
}
__device__ __forceinline__ unsigned lds_off(const void* p) {
  return (unsigned)(size_t)p;
}

// A fragment 16x32 (MxK), bf16, from row-major tile (origin p0, leading dim ld).
// Lanes 0-15: row=lane, K 0..7 (V0..3) and 16..23 (V4..7); lanes 16-31: K 8..15 / 24..31.
__device__ __forceinline__ bhx16 load_a(const bh* p0, int ld, int lane) {
  int row  = lane & 15;
  int koff = (lane >> 4) * 8;
  const bh* p = p0 + (size_t)row * ld + koff;
  bhx8 lo = *(const bhx8*)(p);
  bhx8 hi = *(const bhx8*)(p + 16);
  return cat8(lo, hi);
}

// B fragment 32x16 (KxN) taken from B^T stored row-major [N][K] (origin p0, ld).
// Lane n = lane&15 holds column n: lanes 0-15 K 0..7 / 16..23; lanes 16-31 K 8..15 / 24..31.
__device__ __forceinline__ bhx16 load_bt(const bh* p0, int ld, int lane) {
  int n    = lane & 15;
  int koff = (lane >> 4) * 8;
  const bh* p = p0 + (size_t)n * ld + koff;
  bhx8 lo = *(const bhx8*)(p);
  bhx8 hi = *(const bhx8*)(p + 16);
  return cat8(lo, hi);
}

__device__ __forceinline__ fx8 wmma_bf16(bhx16 a, bhx16 b, fx8 c) {
  return __builtin_amdgcn_wmma_f32_16x16x32_bf16(false, a, false, b, (short)0, c,
                                                 false, false);
}

// ---------------------------------------------------------------------------
// fp32 -> bf16 conversion (inputs and weights), 4 elems/thread
// ---------------------------------------------------------------------------
__global__ __launch_bounds__(256) void cvt_f32_bf16_kernel(
    const float* __restrict__ src, bh* __restrict__ dst, int n)
{
  int i = (blockIdx.x * 256 + threadIdx.x) * 4;
  if (i < n) {
    float4 v = *(const float4*)(src + i);
    bhx4 o;
    o[0] = (bh)v.x; o[1] = (bh)v.y; o[2] = (bh)v.z; o[3] = (bh)v.w;
    *(bhx4*)(dst + i) = o;
  }
}

// ---------------------------------------------------------------------------
// GEMM: out = A[M,1024] @ W^T[N=1024,K=1024] (+bias), WMMA bf16, f32 accum.
// A and W K-slabs (128x32 each) staged into LDS via async DMA once per WG.
// mode 0: head-split bf16 out [B,H,S,Dk], scaled (Q uses 1/sqrt(Dk))
// mode 1: head-split TRANSPOSED bf16 out [B,H,Dk,S]  (for V)
// mode 2: fp32 out[M,1024] = acc + bias + residual    (O projection)
// Workgroup: 8 waves as 2x4; wave tile 64x32 (4x2 WMMA tiles). WG tile 128x128.
// ---------------------------------------------------------------------------
__global__ __launch_bounds__(256) void gemm_proj_kernel(
    const bh* __restrict__ A, const bh* __restrict__ Wt,
    const float* __restrict__ bias, const float* __restrict__ residual,
    bh* __restrict__ outb, float* __restrict__ outf,
    float scale, int mode)
{
  __shared__ bh Abuf[128 * 32];
  __shared__ bh Wbuf[128 * 32];

  int tid  = threadIdx.x;
  int lane = tid & 31;
  int wid  = tid >> 5;
  int m_wg = blockIdx.y * 128;
  int n_wg = blockIdx.x * 128;
  int wm   = (wid >> 2) * 64;
  int wn   = (wid & 3) * 32;

  // async-copy assignment: each thread moves 16 elems (2x B128) per slab
  int crow = tid >> 1;            // 0..127
  int cseg = (tid & 1) * 16;      // element offset 0 or 16

  fx8 acc[4][2];
  #pragma unroll
  for (int i = 0; i < 4; ++i) { acc[i][0] = fzero8(); acc[i][1] = fzero8(); }

  for (int k0 = 0; k0 < DMODEL; k0 += 32) {
    const bh* asrc = A  + (size_t)(m_wg + crow) * DMODEL + k0 + cseg;
    const bh* wsrc = Wt + (size_t)(n_wg + crow) * DMODEL + k0 + cseg;
    unsigned adst = lds_off(&Abuf[crow * 32 + cseg]);
    unsigned wdst = lds_off(&Wbuf[crow * 32 + cseg]);
    async_copy_b128(adst,      asrc);
    async_copy_b128(adst + 16, asrc + 8);
    async_copy_b128(wdst,      wsrc);
    async_copy_b128(wdst + 16, wsrc + 8);
    if (k0 + 32 < DMODEL) {
      __builtin_prefetch(A  + (size_t)(m_wg + crow) * DMODEL + k0 + 32, 0, 3);
      __builtin_prefetch(Wt + (size_t)(n_wg + crow) * DMODEL + k0 + 32, 0, 3);
    }
    wait_async0();
    __syncthreads();

    bhx16 af[4], bfr[2];
    #pragma unroll
    for (int i = 0; i < 4; ++i)
      af[i] = load_a(&Abuf[(wm + 16 * i) * 32], 32, lane);
    #pragma unroll
    for (int j = 0; j < 2; ++j)
      bfr[j] = load_bt(&Wbuf[(wn + 16 * j) * 32], 32, lane);

    #pragma unroll
    for (int i = 0; i < 4; ++i)
      #pragma unroll
      for (int j = 0; j < 2; ++j)
        acc[i][j] = wmma_bf16(af[i], bfr[j], acc[i][j]);

    __syncthreads();  // protect slabs before next iteration's async writes
  }

  int half = lane >> 4, nn = lane & 15;
  #pragma unroll
  for (int i = 0; i < 4; ++i)
    #pragma unroll
    for (int j = 0; j < 2; ++j)
      #pragma unroll
      for (int r = 0; r < 8; ++r) {
        int m = m_wg + wm + 16 * i + r + 8 * half;
        int e = n_wg + wn + 16 * j + nn;
        float v = acc[i][j][r] + bias[e];
        if (mode == 0) {
          v *= scale;
          int b = m >> 11, s = m & (SEQ - 1);
          int h = e >> 6,  d = e & (DHEAD - 1);
          outb[((size_t)(b * NHEAD + h) * SEQ + s) * DHEAD + d] = (bh)v;
        } else if (mode == 1) {
          int b = m >> 11, s = m & (SEQ - 1);
          int h = e >> 6,  d = e & (DHEAD - 1);
          outb[((size_t)(b * NHEAD + h) * DHEAD + d) * SEQ + s] = (bh)v;
        } else {
          size_t idx = (size_t)m * DMODEL + e;
          outf[idx] = v + residual[idx];
        }
      }
}

// ---------------------------------------------------------------------------
// Flash attention: per wave, 16 query rows x full Dk=64, online softmax.
// The 32-key K tile (32x64) and V^T tile (64x32) are shared by all 8 waves and
// staged through LDS with async DMA. P tile (C-layout f32) is converted to
// bf16 through LDS to become an A-fragment for P @ V.
// ---------------------------------------------------------------------------
__global__ __launch_bounds__(256) void attention_kernel(
    const bh* __restrict__ Qh,    // [B*H, S, Dk]  (pre-scaled by 1/sqrt(Dk))
    const bh* __restrict__ Kh,    // [B*H, S, Dk]
    const bh* __restrict__ Vt,    // [B*H, Dk, S]
    const int* __restrict__ mask, // [B, S, S]
    bh* __restrict__ ctx)         // [B, S, DMODEL] merged heads
{
  __shared__ bh Kbuf[32 * 64];       // 32 keys x 64 dk
  __shared__ bh Vbuf[64 * 32];       // 64 dk x 32 keys (from transposed V)
  __shared__ bh Pbuf[8 * 16 * 32];   // per-wave 16x32 prob tile

  int tid  = threadIdx.x;
  int lane = tid & 31;
  int wid  = tid >> 5;
  int bh_i = blockIdx.y;
  int b = bh_i >> 4, h = bh_i & 15;
  int q0 = blockIdx.x * 128 + wid * 16;

  const bh*  Qp = Qh + (size_t)bh_i * SEQ * DHEAD;
  const bh*  Kp = Kh + (size_t)bh_i * SEQ * DHEAD;
  const bh*  Vp = Vt + (size_t)bh_i * DHEAD * SEQ;
  const int* mp = mask + (size_t)b * SEQ * SEQ;
  bh* Ps = &Pbuf[wid * 512];

  // async-copy assignment: K slab 32x64 (8 B128 per row), V slab 64x32 (4 per row)
  int krow = tid >> 3, kseg = (tid & 7) * 8;
  int vrow = tid >> 2, vseg = (tid & 3) * 8;

  bhx16 qa0 = load_a(Qp + (size_t)q0 * DHEAD + 0,  DHEAD, lane);
  bhx16 qa1 = load_a(Qp + (size_t)q0 * DHEAD + 32, DHEAD, lane);

  fx8 cacc[4];
  #pragma unroll
  for (int t = 0; t < 4; ++t) cacc[t] = fzero8();
  float row_max[8], row_sum[8];
  #pragma unroll
  for (int r = 0; r < 8; ++r) { row_max[r] = -1e30f; row_sum[r] = 0.f; }

  int half = lane >> 4, nn = lane & 15;

  for (int kt = 0; kt < SEQ; kt += 32) {
    async_copy_b128(lds_off(&Kbuf[krow * 64 + kseg]),
                    Kp + (size_t)(kt + krow) * DHEAD + kseg);
    async_copy_b128(lds_off(&Vbuf[vrow * 32 + vseg]),
                    Vp + (size_t)vrow * SEQ + kt + vseg);
    wait_async0();
    __syncthreads();   // barrier(1): slabs visible

    // scores: two 16x16 tiles over keys kt..kt+31 (K frags from LDS)
    bhx16 kb00 = load_bt(&Kbuf[0 * 64 + 0],  64, lane);
    bhx16 kb01 = load_bt(&Kbuf[0 * 64 + 32], 64, lane);
    bhx16 kb10 = load_bt(&Kbuf[16 * 64 + 0],  64, lane);
    bhx16 kb11 = load_bt(&Kbuf[16 * 64 + 32], 64, lane);
    fx8 s0 = fzero8(), s1 = fzero8();
    s0 = wmma_bf16(qa0, kb00, s0); s0 = wmma_bf16(qa1, kb01, s0);
    s1 = wmma_bf16(qa0, kb10, s1); s1 = wmma_bf16(qa1, kb11, s1);

    // V fragments loaded BEFORE the P barrier so barrier(2) also protects slabs
    bhx16 vb[4];
    #pragma unroll
    for (int t = 0; t < 4; ++t)
      vb[t] = load_bt(&Vbuf[(t * 16) * 32], 32, lane);

    // mask + online softmax; write probs (bf16) to LDS in row-major [16][32]
    #pragma unroll
    for (int r = 0; r < 8; ++r) {
      int qrow = q0 + r + 8 * half;
      const int* mrow = mp + (size_t)qrow * SEQ + kt;
      float v0 = (mrow[nn]      != 0) ? s0[r] : -1e9f;
      float v1 = (mrow[16 + nn] != 0) ? s1[r] : -1e9f;
      float tmax = fmaxf(v0, v1);
      #pragma unroll
      for (int off = 1; off < 16; off <<= 1)
        tmax = fmaxf(tmax, __shfl_xor(tmax, off, 16));
      float nmax = fmaxf(row_max[r], tmax);
      float corr = __expf(row_max[r] - nmax);
      float p0 = __expf(v0 - nmax);
      float p1 = __expf(v1 - nmax);
      float ps = p0 + p1;
      #pragma unroll
      for (int off = 1; off < 16; off <<= 1)
        ps += __shfl_xor(ps, off, 16);
      row_sum[r] = row_sum[r] * corr + ps;
      row_max[r] = nmax;
      #pragma unroll
      for (int t = 0; t < 4; ++t) cacc[t][r] *= corr;
      int prow = r + 8 * half;
      Ps[prow * 32 + nn]      = (bh)p0;
      Ps[prow * 32 + 16 + nn] = (bh)p1;
    }
    __syncthreads();   // barrier(2): P visible; all slab reads complete

    // reload probs as A-fragment (16x32) from LDS
    bhx16 pa;
    {
      int row  = lane & 15;
      int koff = half * 8;
      #pragma unroll
      for (int i = 0; i < 8; ++i) {
        pa[i]     = Ps[row * 32 + koff + i];
        pa[i + 8] = Ps[row * 32 + koff + 16 + i];
      }
    }

    // ctx += P @ V
    #pragma unroll
    for (int t = 0; t < 4; ++t)
      cacc[t] = wmma_bf16(pa, vb[t], cacc[t]);
  }

  // normalize and write merged-head context (bf16)
  #pragma unroll
  for (int r = 0; r < 8; ++r) {
    int m = q0 + r + 8 * half;
    float inv = 1.0f / row_sum[r];
    #pragma unroll
    for (int t = 0; t < 4; ++t) {
      ctx[((size_t)b * SEQ + m) * DMODEL + h * DHEAD + t * 16 + nn] =
          (bh)(cacc[t][r] * inv);
    }
  }
}

// ---------------------------------------------------------------------------
// LayerNorm over D=1024 per row, in-place on fp32 output.
// ---------------------------------------------------------------------------
__global__ __launch_bounds__(256) void layernorm_kernel(
    float* __restrict__ out, const float* __restrict__ gamma,
    const float* __restrict__ beta)
{
  __shared__ float red[256];
  int row = blockIdx.x;
  float* p = out + (size_t)row * DMODEL;
  int c0 = threadIdx.x * 4;
  float4 x = *(const float4*)(p + c0);

  float s = x.x + x.y + x.z + x.w;
  red[threadIdx.x] = s;
  __syncthreads();
  for (int off = 128; off > 0; off >>= 1) {
    if (threadIdx.x < off) red[threadIdx.x] += red[threadIdx.x + off];
    __syncthreads();
  }
  float mu = red[0] * (1.0f / DMODEL);
  __syncthreads();

  float dx = x.x - mu, dy = x.y - mu, dz = x.z - mu, dw = x.w - mu;
  red[threadIdx.x] = dx*dx + dy*dy + dz*dz + dw*dw;
  __syncthreads();
  for (int off = 128; off > 0; off >>= 1) {
    if (threadIdx.x < off) red[threadIdx.x] += red[threadIdx.x + off];
    __syncthreads();
  }
  float var = red[0] * (1.0f / DMODEL);
  float inv = rsqrtf(var + 1e-5f);

  float4 g = *(const float4*)(gamma + c0);
  float4 bb = *(const float4*)(beta + c0);
  float4 y;
  y.x = dx * inv * g.x + bb.x;
  y.y = dy * inv * g.y + bb.y;
  y.z = dz * inv * g.z + bb.z;
  y.w = dw * inv * g.w + bb.w;
  *(float4*)(p + c0) = y;
}

// ---------------------------------------------------------------------------
extern "C" void kernel_launch(void* const* d_in, const int* in_sizes, int n_in,
                              void* d_out, int out_size, void* d_ws, size_t ws_size,
                              hipStream_t stream) {
  const float* q    = (const float*)d_in[0];
  const float* k    = (const float*)d_in[1];
  const float* v    = (const float*)d_in[2];
  const int*   mask = (const int*)  d_in[3];
  const float* w_q  = (const float*)d_in[4];
  const float* b_q  = (const float*)d_in[5];
  const float* w_k  = (const float*)d_in[6];
  const float* b_k  = (const float*)d_in[7];
  const float* w_v  = (const float*)d_in[8];
  const float* b_v  = (const float*)d_in[9];
  const float* w_o  = (const float*)d_in[10];
  const float* b_o  = (const float*)d_in[11];
  const float* ln_g = (const float*)d_in[12];
  const float* ln_b = (const float*)d_in[13];
  float* out = (float*)d_out;

  char* ws = (char*)d_ws;
  const size_t MB = 1024u * 1024u;
  bh* xq  = (bh*)(ws + 0);          // 8 MB  [4096,1024] bf16
  bh* xk  = (bh*)(ws + 8  * MB);
  bh* xv  = (bh*)(ws + 16 * MB);
  bh* wqb = (bh*)(ws + 24 * MB);    // 2 MB each [1024,1024] bf16
  bh* wkb = (bh*)(ws + 26 * MB);
  bh* wvb = (bh*)(ws + 28 * MB);
  bh* wob = (bh*)(ws + 30 * MB);
  bh* Qh  = (bh*)(ws + 32 * MB);    // 8 MB [B,H,S,Dk]
  bh* Kh  = (bh*)(ws + 40 * MB);
  bh* Vt  = (bh*)(ws + 48 * MB);    // 8 MB [B,H,Dk,S]
  bh* ctx = xq;                      // reuse (xq dead after Q projection)

  dim3 blk(256);
  const int NE_X = MTOT * DMODEL;        // 4194304
  const int NE_W = DMODEL * DMODEL;      // 1048576

  cvt_f32_bf16_kernel<<<NE_X / 1024, blk, 0, stream>>>(q, xq, NE_X);
  cvt_f32_bf16_kernel<<<NE_X / 1024, blk, 0, stream>>>(k, xk, NE_X);
  cvt_f32_bf16_kernel<<<NE_X / 1024, blk, 0, stream>>>(v, xv, NE_X);
  cvt_f32_bf16_kernel<<<NE_W / 1024, blk, 0, stream>>>(w_q, wqb, NE_W);
  cvt_f32_bf16_kernel<<<NE_W / 1024, blk, 0, stream>>>(w_k, wkb, NE_W);
  cvt_f32_bf16_kernel<<<NE_W / 1024, blk, 0, stream>>>(w_v, wvb, NE_W);
  cvt_f32_bf16_kernel<<<NE_W / 1024, blk, 0, stream>>>(w_o, wob, NE_W);

  dim3 gGemm(DMODEL / 128, MTOT / 128);  // (8, 32)
  gemm_proj_kernel<<<gGemm, blk, 0, stream>>>(xq, wqb, b_q, nullptr, Qh, nullptr,
                                              0.125f, 0);  // 1/sqrt(64)
  gemm_proj_kernel<<<gGemm, blk, 0, stream>>>(xk, wkb, b_k, nullptr, Kh, nullptr,
                                              1.0f, 0);
  gemm_proj_kernel<<<gGemm, blk, 0, stream>>>(xv, wvb, b_v, nullptr, Vt, nullptr,
                                              1.0f, 1);

  dim3 gAtt(SEQ / 128, BATCH * NHEAD);   // (16, 32)
  attention_kernel<<<gAtt, blk, 0, stream>>>(Qh, Kh, Vt, mask, ctx);

  gemm_proj_kernel<<<gGemm, blk, 0, stream>>>(ctx, wob, b_o, q, nullptr, out,
                                              1.0f, 2);

  layernorm_kernel<<<MTOT, blk, 0, stream>>>(out, ln_g, ln_b);
}
